// Decoder_86406152061014
// MI455X (gfx1250) — compile-verified, matrix operand order checked by reading, not run on hardware
//
#include <hip/hip_runtime.h>
#include <hip/hip_bf16.h>
#include <cstdint>
#include <cstddef>

// ---------------- problem constants ----------------
#define B_    64
#define S_    100
#define T_    16
#define H_    8
#define DH_   384
#define D_    512
#define E_    512
#define WV_   512
#define V_    50000
#define GIN_  1024          // WV + E
#define HDH_  3072          // H * DH
#define MO_   256           // D / POOL
#define UNK_  1

typedef __bf16 bf16;
typedef __attribute__((ext_vector_type(16))) __bf16 v16bf;
typedef __attribute__((ext_vector_type(8)))  __bf16 v8bf;
typedef __attribute__((ext_vector_type(8)))  float  v8f;

// ---------------- output offsets (floats), reference tuple order ----------------
static const size_t O0  = 0;                         // sample_y (T,B)
static const size_t O1  = 1024;                      // wordLks (T,B,V)
static const size_t O2  = 51201024;                  // copyLks (T,B,S)
static const size_t O3  = 51303424;                  // copyPs  (T,B)
static const size_t O4  = 51304448;                  // hF      (B,D)
static const size_t O5  = 51337216;                  // c_attn last (B,S)
static const size_t O6  = 51343616;                  // curF    (B,E)
static const size_t O7  = 51376384;                  // mul_as  (T,B,H,S)
static const size_t O8  = 52195584;                  // isCs    (T,B)
static const size_t O9  = 52196608;                  // preds   (T,B)
static const size_t O10 = 52197632;                  // mul_cs  (T,B,H*DH)
static const size_t O11 = 55343360;                  // mul_as again

// ---------------- helpers ----------------
__device__ __forceinline__ bf16 f2bf(float f) {
  unsigned u = __builtin_bit_cast(unsigned, f);
  unsigned r = u + 0x7FFFu + ((u >> 16) & 1u);     // round-to-nearest-even
  unsigned short h = (unsigned short)(r >> 16);
  return __builtin_bit_cast(bf16, h);
}
__device__ __forceinline__ float bf2f(bf16 b) {
  unsigned short h = __builtin_bit_cast(unsigned short, b);
  unsigned u = ((unsigned)h) << 16;
  return __builtin_bit_cast(float, u);
}

// ---------------- WMMA bf16 GEMM: C[M,N] = A[M,K] * W[N,K]^T + bias ----------------
// One wave owns a 64x16 output strip (4 M-tiles x 1 N-tile): the B (weight) tile is
// loaded ONCE per K-step and reused by 4 independent WMMAs -> 4x less weight traffic
// on the memory-bound generator GEMM, and hazard-free back-to-back WMMA issue.
// Prefetch is UNCONDITIONAL: speculative prefetch past the tensor end is dropped by
// hardware (ISA 10.5), and keeping the loop branch-free lets the scheduler pipeline
// loads across WMMAs (partial s_wait_loadcnt instead of 0x0).
// Requires M % 64 == 0, N % 16 == 0, K % 32 == 0 (true for all call sites).
union V16U { v16bf v; v8bf h[2]; };

__global__ void k_gemm(const bf16* __restrict__ A, const bf16* __restrict__ W,
                       const float* __restrict__ bias,
                       float* __restrict__ Cf, bf16* __restrict__ Cb,
                       int M, int N, int K) {
  const int wave  = (int)((blockIdx.x * blockDim.x + threadIdx.x) >> 5);
  const int lane  = threadIdx.x & 31;
  const int tilesN  = N >> 4;
  const int mBlocks = M >> 6;                       // 64-row strips
  const int bm = wave / tilesN;
  const int tn = wave - bm * tilesN;
  if (bm >= mBlocks) return;                        // wave-uniform exit (EXEC all-1 for WMMA)
  const int half = lane >> 4;
  const int l15  = lane & 15;
  // 16-bit A layout: lanes 0-15 hold K=0..7,16..23 ; lanes 16-31 hold K=8..15,24..31
  const bf16* a0   = A + (size_t)(bm * 64 + l15) * K + half * 8;
  const bf16* wrow = W + (size_t)(tn * 16 + l15) * K + half * 8;  // column n of B == row n of W
  const size_t rstep = (size_t)16 * K;              // one M-tile down
  v8f acc0 = {}, acc1 = {}, acc2 = {}, acc3 = {};
  for (int k0 = 0; k0 < K; k0 += 32) {
    __builtin_prefetch((const void*)(wrow + k0 + 128), 0, 2);   // speculative stream-ahead
    V16U b;
    b.h[0] = *(const v8bf*)(wrow + k0);
    b.h[1] = *(const v8bf*)(wrow + k0 + 16);
    V16U t0, t1, t2, t3;
    t0.h[0] = *(const v8bf*)(a0 + k0);
    t0.h[1] = *(const v8bf*)(a0 + k0 + 16);
    t1.h[0] = *(const v8bf*)(a0 + rstep + k0);
    t1.h[1] = *(const v8bf*)(a0 + rstep + k0 + 16);
    t2.h[0] = *(const v8bf*)(a0 + 2 * rstep + k0);
    t2.h[1] = *(const v8bf*)(a0 + 2 * rstep + k0 + 16);
    t3.h[0] = *(const v8bf*)(a0 + 3 * rstep + k0);
    t3.h[1] = *(const v8bf*)(a0 + 3 * rstep + k0 + 16);
    acc0 = __builtin_amdgcn_wmma_f32_16x16x32_bf16(false, t0.v, false, b.v, (short)0, acc0, false, false);
    acc1 = __builtin_amdgcn_wmma_f32_16x16x32_bf16(false, t1.v, false, b.v, (short)0, acc1, false, false);
    acc2 = __builtin_amdgcn_wmma_f32_16x16x32_bf16(false, t2.v, false, b.v, (short)0, acc2, false, false);
    acc3 = __builtin_amdgcn_wmma_f32_16x16x32_bf16(false, t3.v, false, b.v, (short)0, acc3, false, false);
  }
  const int n = tn * 16 + l15;
  const float bsv = bias ? bias[n] : 0.0f;
  v8f accs[4] = {acc0, acc1, acc2, acc3};
#pragma unroll
  for (int i = 0; i < 4; ++i) {
#pragma unroll
    for (int r = 0; r < 8; ++r) {
      const int m = bm * 64 + i * 16 + r + half * 8;  // D layout: VGPR r -> row r (+8 hi lanes)
      float val = accs[i][r] + bsv;
      if (Cf) Cf[(size_t)m * N + n] = val;
      if (Cb) Cb[(size_t)m * N + n] = f2bf(val);
    }
  }
}

// ---------------- small utility kernels ----------------
__global__ void k_cvt_bf16(const float* __restrict__ s, bf16* __restrict__ d, int n) {
  for (int i = blockIdx.x * blockDim.x + threadIdx.x; i < n; i += gridDim.x * blockDim.x)
    d[i] = f2bf(s[i]);
}
__global__ void k_copy_f32(const float* __restrict__ s, float* __restrict__ d, int n) {
  for (int i = blockIdx.x * blockDim.x + threadIdx.x; i < n; i += gridDim.x * blockDim.x)
    d[i] = s[i];
}
__global__ void k_init_tok(const int* __restrict__ toks, int* __restrict__ tok) {
  int b = threadIdx.x;
  if (b < B_) tok[b] = toks[b];                     // input_tokens[0,0,b]
}
// context (S,B,E) -> row-major (B*S, E) bf16
__global__ void k_build_ctx(const float* __restrict__ ctx, bf16* __restrict__ dst) {
  int i = blockIdx.x * blockDim.x + threadIdx.x;
  if (i >= B_ * S_ * E_) return;
  int e = i % E_; int bs = i / E_; int b = bs / S_; int s = bs - b * S_;
  dst[i] = f2bf(ctx[((size_t)s * B_ + b) * E_ + e]);
}
// x = concat(emb(tok), cur) -> bf16 (B, GIN)
__global__ void k_build_x(const float* __restrict__ lut, const float* __restrict__ cur,
                          const int* __restrict__ tok, bf16* __restrict__ x) {
  int i = blockIdx.x * blockDim.x + threadIdx.x;
  if (i >= B_ * GIN_) return;
  int j = i & (GIN_ - 1); int b = i >> 10;
  float v = (j < WV_) ? lut[(size_t)tok[b] * WV_ + j] : cur[b * E_ + (j - WV_)];
  x[i] = f2bf(v);
}
__global__ void k_gru(const float* __restrict__ gi, const float* __restrict__ gh,
                      float* __restrict__ h, bf16* __restrict__ h_bf,
                      float* __restrict__ out) {
  int i = blockIdx.x * blockDim.x + threadIdx.x;
  if (i >= B_ * D_) return;
  int b = i / D_, d = i - b * D_;
  const float* gib = gi + (size_t)b * 3 * D_;
  const float* ghb = gh + (size_t)b * 3 * D_;
  float r = 1.0f / (1.0f + expf(-(gib[d] + ghb[d])));
  float z = 1.0f / (1.0f + expf(-(gib[D_ + d] + ghb[D_ + d])));
  float n = tanhf(gib[2 * D_ + d] + r * ghb[2 * D_ + d]);
  float h0 = (1.0f - z) * n + z * h[i];
  h[i] = h0; h_bf[i] = f2bf(h0);
  out[O4 + i] = h0;                                 // hF (overwritten each step; last wins)
}
// scores: q (f32) dot K rows (bf16, contiguous along d) -- vectorized b128 loads
__global__ void k_scores(const float* __restrict__ q, const bf16* __restrict__ kbf,
                         const unsigned char* __restrict__ mask, float* __restrict__ sc) {
  int i = blockIdx.x * blockDim.x + threadIdx.x;
  if (i >= B_ * H_ * S_) return;
  int s = i % S_; int bh = i / S_; int h = bh % H_; int b = bh / H_;
  const float* qp = q + (size_t)b * HDH_ + h * DH_;
  const bf16*  kp = kbf + ((size_t)b * S_ + s) * HDH_ + h * DH_;
  float acc = 0.0f;
  for (int d = 0; d < DH_; d += 8) {
    v8bf kv = *(const v8bf*)(kp + d);
#pragma unroll
    for (int j = 0; j < 8; ++j) acc += qp[d + j] * bf2f(kv[j]);
  }
  acc *= 0.051031036307982884f;                     // 1/sqrt(384)
  if (mask[b * S_ + s]) acc = -1.0e9f;
  sc[i] = acc;
}
__global__ void k_softmax(const float* __restrict__ sc, float* __restrict__ attn,
                          float* __restrict__ out, int t) {
  int wave = (int)((blockIdx.x * blockDim.x + threadIdx.x) >> 5);
  int lane = threadIdx.x & 31;
  if (wave >= B_ * H_) return;
  const float* row = sc + (size_t)wave * S_;
  float v[4]; float mx = -1.0e30f;
#pragma unroll
  for (int k = 0; k < 4; ++k) { int s = lane + 32 * k; v[k] = (s < S_) ? row[s] : -1.0e30f; mx = fmaxf(mx, v[k]); }
  for (int o = 16; o > 0; o >>= 1) mx = fmaxf(mx, __shfl_xor(mx, o, 32));
  float sum = 0.0f;
#pragma unroll
  for (int k = 0; k < 4; ++k) { int s = lane + 32 * k; if (s < S_) { v[k] = expf(v[k] - mx); sum += v[k]; } else v[k] = 0.0f; }
  for (int o = 16; o > 0; o >>= 1) sum += __shfl_xor(sum, o, 32);
  float inv = 1.0f / sum;
  size_t b7  = O7  + (size_t)t * B_ * H_ * S_ + (size_t)wave * S_;
  size_t b11 = O11 + (size_t)t * B_ * H_ * S_ + (size_t)wave * S_;
#pragma unroll
  for (int k = 0; k < 4; ++k) {
    int s = lane + 32 * k;
    if (s < S_) { float a = v[k] * inv; attn[(size_t)wave * S_ + s] = a; out[b7 + s] = a; out[b11 + s] = a; }
  }
}
__global__ void k_cattn(const float* __restrict__ attn, float* __restrict__ ca,
                        float* __restrict__ out) {
  int i = blockIdx.x * blockDim.x + threadIdx.x;
  if (i >= B_ * S_) return;
  int b = i / S_, s = i - b * S_;
  float acc = 0.0f;
#pragma unroll
  for (int h = 0; h < H_; ++h) acc += attn[((size_t)b * H_ + h) * S_ + s];
  acc *= 0.125f;
  ca[i] = acc; out[O5 + i] = acc;                   // last step's value survives
}
// mul_c: attn-weighted sum of V rows; each thread owns 8 consecutive cols -> b128 V loads
__global__ void k_mulc(const float* __restrict__ attn, const bf16* __restrict__ vbf,
                       bf16* __restrict__ mulc_bf, float* __restrict__ out, int t) {
  int i = blockIdx.x * blockDim.x + threadIdx.x;
  if (i >= B_ * (HDH_ / 8)) return;
  int g = i % (HDH_ / 8); int b = i / (HDH_ / 8);
  int col = g * 8; int h = col / DH_;               // 8-col group never crosses a head (384 % 8 == 0)
  const bf16*  vp = vbf + (size_t)b * S_ * HDH_ + col;
  const float* ap = attn + ((size_t)b * H_ + h) * S_;
  float acc[8] = {};
  for (int s = 0; s < S_; ++s) {
    v8bf vv = *(const v8bf*)(vp + (size_t)s * HDH_);
    float w = ap[s];
#pragma unroll
    for (int j = 0; j < 8; ++j) acc[j] += w * bf2f(vv[j]);
  }
  size_t ob = O10 + (size_t)t * B_ * HDH_ + (size_t)b * HDH_ + col;
#pragma unroll
  for (int j = 0; j < 8; ++j) {
    out[ob + j] = acc[j];
    mulc_bf[(size_t)b * HDH_ + col + j] = f2bf(acc[j]);
  }
}
__global__ void k_copyp(const float* __restrict__ h, const float* __restrict__ cur2,
                        const float* __restrict__ Wc, const float* __restrict__ bc,
                        float* __restrict__ cp, float* __restrict__ out, int t) {
  int wave = (int)((blockIdx.x * blockDim.x + threadIdx.x) >> 5);
  int lane = threadIdx.x & 31;
  if (wave >= B_) return;
  float acc = 0.0f;
  for (int j = lane; j < D_ + E_; j += 32) {
    float xv = (j < D_) ? h[wave * D_ + j] : cur2[wave * E_ + (j - D_)];
    acc += xv * Wc[j];
  }
  for (int o = 16; o > 0; o >>= 1) acc += __shfl_xor(acc, o, 32);
  if (lane == 0) {
    float p = 1.0f / (1.0f + expf(-(acc + bc[0])));
    cp[wave] = p; out[O3 + (size_t)t * B_ + wave] = p;
  }
}
__global__ void k_roin(const float* __restrict__ lut, const int* __restrict__ tok,
                       const float* __restrict__ h, const float* __restrict__ cur2,
                       bf16* __restrict__ ro_in) {
  int i = blockIdx.x * blockDim.x + threadIdx.x;
  if (i >= B_ * 1536) return;
  int j = i % 1536; int b = i / 1536;
  float v = (j < WV_) ? lut[(size_t)tok[b] * WV_ + j]
          : (j < WV_ + D_) ? h[b * D_ + (j - WV_)]
          : cur2[b * E_ + (j - WV_ - D_)];
  ro_in[i] = f2bf(v);
}
__global__ void k_maxpool(const float* __restrict__ ro, bf16* __restrict__ mo) {
  int i = blockIdx.x * blockDim.x + threadIdx.x;
  if (i >= B_ * MO_) return;
  int b = i / MO_, j = i - b * MO_;
  mo[i] = f2bf(fmaxf(ro[b * D_ + 2 * j], ro[b * D_ + 2 * j + 1]));
}
__global__ void k_wordlk(const float* __restrict__ lg, const float* __restrict__ cp,
                         float* __restrict__ out, int t) {
  __shared__ float sh[256];
  int b = blockIdx.x, tid = threadIdx.x;
  const float* l = lg + (size_t)b * V_;
  float mx = -1.0e30f;
  for (int v = tid; v < V_; v += 256) mx = fmaxf(mx, l[v]);
  sh[tid] = mx; __syncthreads();
  for (int s = 128; s > 0; s >>= 1) { if (tid < s) sh[tid] = fmaxf(sh[tid], sh[tid + s]); __syncthreads(); }
  mx = sh[0]; __syncthreads();
  float sum = 0.0f;
  for (int v = tid; v < V_; v += 256) sum += expf(l[v] - mx);
  sh[tid] = sum; __syncthreads();
  for (int s = 128; s > 0; s >>= 1) { if (tid < s) sh[tid] += sh[tid + s]; __syncthreads(); }
  sum = sh[0];
  float inv = 1.0f / sum, om = 1.0f - cp[b];
  size_t base = O1 + (size_t)t * B_ * V_ + (size_t)b * V_;
  for (int v = tid; v < V_; v += 256)
    out[base + v] = logf(expf(l[v] - mx) * inv * om + 1e-8f);
}
__global__ void k_copylk(const float* __restrict__ ca, const float* __restrict__ cp,
                         float* __restrict__ out, int t) {
  int i = blockIdx.x * blockDim.x + threadIdx.x;
  if (i >= B_ * S_) return;
  int b = i / S_;
  out[O2 + (size_t)t * B_ * S_ + i] = logf(ca[i] * cp[b] + 1e-8f);
}
__global__ void k_argmax(float* __restrict__ out, int* __restrict__ tok, int t) {
  __shared__ float sv[256]; __shared__ int si[256];
  int b = blockIdx.x, tid = threadIdx.x;
  float best = -1.0e30f; int bi = 0x7fffffff;
  size_t wbase = O1 + (size_t)t * B_ * V_ + (size_t)b * V_;
  for (int v = tid; v < V_; v += 256) {
    float x = out[wbase + v];
    if (x > best || (x == best && v < bi)) { best = x; bi = v; }
  }
  size_t cbase = O2 + (size_t)t * B_ * S_ + (size_t)b * S_;
  for (int s = tid; s < S_; s += 256) {
    float x = out[cbase + s]; int idx = V_ + s;
    if (x > best || (x == best && idx < bi)) { best = x; bi = idx; }
  }
  sv[tid] = best; si[tid] = bi; __syncthreads();
  for (int s = 128; s > 0; s >>= 1) {
    if (tid < s) {
      if (sv[tid + s] > sv[tid] || (sv[tid + s] == sv[tid] && si[tid + s] < si[tid])) {
        sv[tid] = sv[tid + s]; si[tid] = si[tid + s];
      }
    }
    __syncthreads();
  }
  if (tid == 0) {
    int bidx = si[0]; int isC = bidx >= V_ ? 1 : 0; int fin = isC ? UNK_ : bidx;
    out[O0 + (size_t)t * B_ + b] = (float)fin;
    out[O8 + (size_t)t * B_ + b] = (float)isC;
    out[O9 + (size_t)t * B_ + b] = (float)bidx;
    tok[b] = fin;
  }
}
__global__ void k_curf(const float* __restrict__ cur, float* __restrict__ out) {
  int i = blockIdx.x * blockDim.x + threadIdx.x;
  if (i < B_ * E_) out[O6 + i] = cur[i];
}

// ---------------- host ----------------
extern "C" void kernel_launch(void* const* d_in, const int* in_sizes, int n_in,
                              void* d_out, int out_size, void* d_ws, size_t ws_size,
                              hipStream_t stream) {
  const int*   inTok    = (const int*)d_in[0];
  const float* hidden   = (const float*)d_in[1];
  const float* context  = (const float*)d_in[2];
  const unsigned char* mask = (const unsigned char*)d_in[3];
  const float* init_att = (const float*)d_in[4];
  const float* word_lut = (const float*)d_in[6];
  const float* W_ih = (const float*)d_in[7];
  const float* W_hh = (const float*)d_in[8];
  const float* b_ih = (const float*)d_in[9];
  const float* b_hh = (const float*)d_in[10];
  const float* Wk   = (const float*)d_in[11];
  const float* bk   = (const float*)d_in[12];
  const float* Wv   = (const float*)d_in[13];
  const float* bv   = (const float*)d_in[14];
  const float* Wq   = (const float*)d_in[15];
  const float* bq   = (const float*)d_in[16];
  const float* Wo   = (const float*)d_in[17];
  const float* bo   = (const float*)d_in[18];
  const float* W_copy = (const float*)d_in[19];
  const float* b_copy = (const float*)d_in[20];
  const float* W_read = (const float*)d_in[21];
  const float* b_read = (const float*)d_in[22];
  const float* W_gen  = (const float*)d_in[23];
  const float* b_gen  = (const float*)d_in[24];
  float* out = (float*)d_out;

  // workspace carve-up (256B aligned)
  char* basep = (char*)d_ws; size_t off = 0;
  auto alloc = [&](size_t bytes) -> char* {
    char* p = basep + off; off = (off + bytes + 255) & ~(size_t)255; return p;
  };
  bf16* wih_bf  = (bf16*)alloc((size_t)3 * D_ * GIN_ * 2);
  bf16* whh_bf  = (bf16*)alloc((size_t)3 * D_ * D_ * 2);
  bf16* wq_bf   = (bf16*)alloc((size_t)HDH_ * D_ * 2);
  bf16* wk_bf   = (bf16*)alloc((size_t)HDH_ * D_ * 2);
  bf16* wv_bf   = (bf16*)alloc((size_t)HDH_ * D_ * 2);
  bf16* wo_bf   = (bf16*)alloc((size_t)E_ * HDH_ * 2);
  bf16* wread_bf= (bf16*)alloc((size_t)D_ * 1536 * 2);
  bf16* wgen_bf = (bf16*)alloc((size_t)V_ * MO_ * 2);
  bf16* ctx_bf  = (bf16*)alloc((size_t)B_ * S_ * E_ * 2);
  bf16* kbf     = (bf16*)alloc((size_t)B_ * S_ * HDH_ * 2);
  bf16* vbf     = (bf16*)alloc((size_t)B_ * S_ * HDH_ * 2);
  bf16* x_bf    = (bf16*)alloc((size_t)B_ * GIN_ * 2);
  float* h_f    = (float*)alloc((size_t)B_ * D_ * 4);
  bf16* h_bf    = (bf16*)alloc((size_t)B_ * D_ * 2);
  float* gi     = (float*)alloc((size_t)B_ * 3 * D_ * 4);
  float* gh     = (float*)alloc((size_t)B_ * 3 * D_ * 4);
  float* q_f    = (float*)alloc((size_t)B_ * HDH_ * 4);
  float* scores = (float*)alloc((size_t)B_ * H_ * S_ * 4);
  float* attn   = (float*)alloc((size_t)B_ * H_ * S_ * 4);
  float* cattn  = (float*)alloc((size_t)B_ * S_ * 4);
  bf16* mulc_bf = (bf16*)alloc((size_t)B_ * HDH_ * 2);
  float* cur_f  = (float*)alloc((size_t)B_ * E_ * 4);
  bf16* cur_bf  = (bf16*)alloc((size_t)B_ * E_ * 2);
  float* copyp  = (float*)alloc((size_t)B_ * 4);
  bf16* roin_bf = (bf16*)alloc((size_t)B_ * 1536 * 2);
  float* ro_f   = (float*)alloc((size_t)B_ * D_ * 4);
  bf16* mo_bf   = (bf16*)alloc((size_t)B_ * MO_ * 2);
  float* logits = (float*)alloc((size_t)B_ * V_ * 4);
  int*  tok     = (int*)alloc((size_t)B_ * 4);

  auto cdiv = [](long n, long d) { return (int)((n + d - 1) / d); };
  auto gemm = [&](const bf16* A, const bf16* W, const float* bias,
                  float* Cf, bf16* Cb, int M, int N, int K) {
    int waves = (M / 64) * (N / 16);                // one wave per 64x16 strip
    k_gemm<<<dim3(cdiv(waves, 8)), dim3(256), 0, stream>>>(A, W, bias, Cf, Cb, M, N, K);
  };

  // ---- one-time (per launch) weight conversion + K/V projections ----
  k_cvt_bf16<<<cdiv(3 * D_ * GIN_, 256), 256, 0, stream>>>(W_ih, wih_bf, 3 * D_ * GIN_);
  k_cvt_bf16<<<cdiv(3 * D_ * D_, 256), 256, 0, stream>>>(W_hh, whh_bf, 3 * D_ * D_);
  k_cvt_bf16<<<cdiv(HDH_ * D_, 256), 256, 0, stream>>>(Wq, wq_bf, HDH_ * D_);
  k_cvt_bf16<<<cdiv(HDH_ * D_, 256), 256, 0, stream>>>(Wk, wk_bf, HDH_ * D_);
  k_cvt_bf16<<<cdiv(HDH_ * D_, 256), 256, 0, stream>>>(Wv, wv_bf, HDH_ * D_);
  k_cvt_bf16<<<cdiv(E_ * HDH_, 256), 256, 0, stream>>>(Wo, wo_bf, E_ * HDH_);
  k_cvt_bf16<<<cdiv(D_ * 1536, 256), 256, 0, stream>>>(W_read, wread_bf, D_ * 1536);
  k_cvt_bf16<<<cdiv(V_ * MO_, 256), 256, 0, stream>>>(W_gen, wgen_bf, V_ * MO_);
  k_build_ctx<<<cdiv(B_ * S_ * E_, 256), 256, 0, stream>>>(context, ctx_bf);
  gemm(ctx_bf, wk_bf, bk, nullptr, kbf, B_ * S_, HDH_, E_);   // k_s
  gemm(ctx_bf, wv_bf, bv, nullptr, vbf, B_ * S_, HDH_, E_);   // v_s
  k_copy_f32<<<cdiv(B_ * D_, 256), 256, 0, stream>>>(hidden, h_f, B_ * D_);
  k_cvt_bf16<<<cdiv(B_ * D_, 256), 256, 0, stream>>>(hidden, h_bf, B_ * D_);
  k_copy_f32<<<cdiv(B_ * E_, 256), 256, 0, stream>>>(init_att, cur_f, B_ * E_);
  k_init_tok<<<1, 64, 0, stream>>>(inTok, tok);

  // ---- sequential decode ----
  for (int t = 0; t < T_; ++t) {
    k_build_x<<<cdiv(B_ * GIN_, 256), 256, 0, stream>>>(word_lut, cur_f, tok, x_bf);
    gemm(x_bf, wih_bf, b_ih, gi, nullptr, B_, 3 * D_, GIN_);
    gemm(h_bf, whh_bf, b_hh, gh, nullptr, B_, 3 * D_, D_);
    k_gru<<<cdiv(B_ * D_, 256), 256, 0, stream>>>(gi, gh, h_f, h_bf, out);
    gemm(h_bf, wq_bf, bq, q_f, nullptr, B_, HDH_, D_);
    k_scores<<<cdiv(B_ * H_ * S_, 256), 256, 0, stream>>>(q_f, kbf, mask, scores);
    k_softmax<<<cdiv(B_ * H_ * 32, 256), 256, 0, stream>>>(scores, attn, out, t);
    k_cattn<<<cdiv(B_ * S_, 256), 256, 0, stream>>>(attn, cattn, out);
    k_mulc<<<cdiv(B_ * (HDH_ / 8), 256), 256, 0, stream>>>(attn, vbf, mulc_bf, out, t);
    gemm(mulc_bf, wo_bf, bo, cur_f, cur_bf, B_, E_, HDH_);    // cur2 (reuses cur buffer)
    k_copyp<<<cdiv(B_ * 32, 256), 256, 0, stream>>>(h_f, cur_f, W_copy, b_copy, copyp, out, t);
    k_roin<<<cdiv(B_ * 1536, 256), 256, 0, stream>>>(word_lut, tok, h_f, cur_f, roin_bf);
    gemm(roin_bf, wread_bf, b_read, ro_f, nullptr, B_, D_, 1536);
    k_maxpool<<<cdiv(B_ * MO_, 256), 256, 0, stream>>>(ro_f, mo_bf);
    gemm(mo_bf, wgen_bf, b_gen, logits, nullptr, B_, V_, MO_);
    k_wordlk<<<B_, 256, 0, stream>>>(logits, copyp, out, t);
    k_copylk<<<cdiv(B_ * S_, 256), 256, 0, stream>>>(cattn, copyp, out, t);
    k_argmax<<<B_, 256, 0, stream>>>(out, tok, t);
  }
  k_curf<<<cdiv(B_ * E_, 256), 256, 0, stream>>>(cur_f, out);
}
